// DreamConsolidationEngine_53523882443047
// MI455X (gfx1250) — compile-verified, matrix operand order checked by reading, not run on hardware
//
#include <hip/hip_runtime.h>
#include <hip/hip_bf16.h>

// ---------------------------------------------------------------------------
// Problem dims (fixed by the reference)
// ---------------------------------------------------------------------------
#define HD    1024      // hidden dim H
#define CD    256       // compressed dim C
#define MMEM  100000    // memory size
#define NS    1024      // num samples == GRU sequence length T
#define G3H   3072      // 3*H

typedef __attribute__((ext_vector_type(16))) __bf16 v16bf;
typedef __attribute__((ext_vector_type(8)))  float  v8f;

// ---------------------------------------------------------------------------
// Fragment load: 16 bf16 elements per lane as two 16-byte global loads.
// Per the CDNA5 16-bit A/B fragment layout, lane half (lane>>4) selects the
// K sub-range; elements 0-7 live at k+half*8, elements 8-15 at k+16+half*8.
// ---------------------------------------------------------------------------
__device__ __forceinline__ v16bf load_frag16(const __bf16* p) {
  v16bf f;
  uint4* u = reinterpret_cast<uint4*>(&f);
  u[0] = *reinterpret_cast<const uint4*>(p);
  u[1] = *reinterpret_cast<const uint4*>(p + 16);
  return f;
}

// ---------------------------------------------------------------------------
// WMMA GEMM (NT):  C[M,N] fp32 = A[M,K] bf16 (row major) x B[N,K] bf16
// (row major, i.e. W stored [out,in], computing x @ W^T) + bias[N].
// Block = 256 threads = 8 waves. Block tile 128(M) x 64(N); each wave owns a
// 16x64 strip. K-loop unrolled 2x with two structurally-alternating fragment
// buffer sets: zero register copies, and each buffer's refill loads retire
// during the opposite stage's 4 WMMAs (partial s_wait_loadcnt, no mov pile).
// Requires Kdim % 64 == 0 (holds for all GEMMs here: K in {256, 512, 1024}).
// ---------------------------------------------------------------------------
__global__ __launch_bounds__(256) void gemm_bf16_nt_bias(
    const __bf16* __restrict__ A, const __bf16* __restrict__ B,
    const float* __restrict__ bias, float* __restrict__ Cc,
    int Mdim, int Ndim, int Kdim)
{
  const int wave = threadIdx.x >> 5;
  const int lane = threadIdx.x & 31;
  const int half = lane >> 4;
  const int r    = lane & 15;
  const int m0   = blockIdx.y * 128 + wave * 16;
  const int n0   = blockIdx.x * 64;

  v8f acc[4];
#pragma unroll
  for (int nt = 0; nt < 4; ++nt) {
    const float bv = bias[n0 + nt * 16 + r];   // C col = r, same for all 8 rows
#pragma unroll
    for (int i = 0; i < 8; ++i) acc[nt][i] = bv;
  }

  const __bf16* arow = A + (size_t)(m0 + r) * Kdim + half * 8;
  const __bf16* brow[4];
#pragma unroll
  for (int nt = 0; nt < 4; ++nt)
    brow[nt] = B + (size_t)(n0 + nt * 16 + r) * Kdim + half * 8;

  // prologue: two buffer sets, k=0 and k=32
  v16bf a0 = load_frag16(arow);
  v16bf b0[4];
#pragma unroll
  for (int nt = 0; nt < 4; ++nt) b0[nt] = load_frag16(brow[nt]);
  v16bf a1 = load_frag16(arow + 32);
  v16bf b1[4];
#pragma unroll
  for (int nt = 0; nt < 4; ++nt) b1[nt] = load_frag16(brow[nt] + 32);

  int k0 = 0;
  for (; k0 + 64 < Kdim; k0 += 64) {
    __builtin_prefetch(arow + k0 + 320, 0, 1);          // global_prefetch_b8

    // stage 0: consume buf0 (loaded >= one full stage ago), refill @ k0+64
#pragma unroll
    for (int nt = 0; nt < 4; ++nt)
      acc[nt] = __builtin_amdgcn_wmma_f32_16x16x32_bf16(
          false, a0, false, b0[nt], (short)0, acc[nt], false, false);
    a0 = load_frag16(arow + k0 + 64);
#pragma unroll
    for (int nt = 0; nt < 4; ++nt) b0[nt] = load_frag16(brow[nt] + k0 + 64);

    // stage 1: consume buf1, refill @ k0+96
#pragma unroll
    for (int nt = 0; nt < 4; ++nt)
      acc[nt] = __builtin_amdgcn_wmma_f32_16x16x32_bf16(
          false, a1, false, b1[nt], (short)0, acc[nt], false, false);
    a1 = load_frag16(arow + k0 + 96);
#pragma unroll
    for (int nt = 0; nt < 4; ++nt) b1[nt] = load_frag16(brow[nt] + k0 + 96);
  }

  // epilogue: final two K-steps (no refills)
#pragma unroll
  for (int nt = 0; nt < 4; ++nt)
    acc[nt] = __builtin_amdgcn_wmma_f32_16x16x32_bf16(
        false, a0, false, b0[nt], (short)0, acc[nt], false, false);
#pragma unroll
  for (int nt = 0; nt < 4; ++nt)
    acc[nt] = __builtin_amdgcn_wmma_f32_16x16x32_bf16(
        false, a1, false, b1[nt], (short)0, acc[nt], false, false);

#pragma unroll
  for (int nt = 0; nt < 4; ++nt)
#pragma unroll
    for (int v = 0; v < 8; ++v)
      Cc[(size_t)(m0 + half * 8 + v) * Ndim + n0 + nt * 16 + r] = acc[nt][v];
}

// ---------------------------------------------------------------------------
// fp32 -> bf16 conversion (grid strided)
// ---------------------------------------------------------------------------
__global__ void cvt_f32_bf16(const float* __restrict__ x, __bf16* __restrict__ y, int n)
{
  for (int i = blockIdx.x * blockDim.x + threadIdx.x; i < n;
       i += gridDim.x * blockDim.x)
    y[i] = (__bf16)x[i];
}

// ---------------------------------------------------------------------------
// Gather sampled memories: fp32 copy (for losses) + bf16 copy (for GEMM A)
// ---------------------------------------------------------------------------
__global__ void gather_rows(const float* __restrict__ mem, const int* __restrict__ idx,
                            float* __restrict__ sf, __bf16* __restrict__ sb)
{
  const int row = blockIdx.x;
  const float* p = mem + (size_t)idx[row] * HD;
  for (int j = threadIdx.x; j < HD; j += blockDim.x) {
    float v = p[j];
    sf[(size_t)row * HD + j] = v;
    sb[(size_t)row * HD + j] = (__bf16)v;
  }
}

// ---------------------------------------------------------------------------
// LayerNorm (D=512) + exact GELU, emitting bf16 for the next GEMM.
// One block (256 threads, 2 elems/thread) per row.
// ---------------------------------------------------------------------------
__global__ __launch_bounds__(256) void ln_gelu_512(
    const float* __restrict__ X, const float* __restrict__ g,
    const float* __restrict__ b, __bf16* __restrict__ out)
{
  __shared__ float red[256];
  const int row = blockIdx.x, tid = threadIdx.x;
  const float* xr = X + (size_t)row * 512;
  float x0 = xr[tid], x1 = xr[tid + 256];

  red[tid] = x0 + x1; __syncthreads();
  for (int s = 128; s; s >>= 1) { if (tid < s) red[tid] += red[tid + s]; __syncthreads(); }
  const float mean = red[0] * (1.0f / 512.0f);
  __syncthreads();

  const float d0 = x0 - mean, d1 = x1 - mean;
  red[tid] = d0 * d0 + d1 * d1; __syncthreads();
  for (int s = 128; s; s >>= 1) { if (tid < s) red[tid] += red[tid + s]; __syncthreads(); }
  const float rstd = rsqrtf(red[0] * (1.0f / 512.0f) + 1e-5f);

  float y0 = d0 * rstd * g[tid]       + b[tid];
  float y1 = d1 * rstd * g[tid + 256] + b[tid + 256];
  y0 = 0.5f * y0 * (1.0f + erff(y0 * 0.70710678118654752f));
  y1 = 0.5f * y1 * (1.0f + erff(y1 * 0.70710678118654752f));
  out[(size_t)row * 512 + tid]       = (__bf16)y0;
  out[(size_t)row * 512 + tid + 256] = (__bf16)y1;
}

// ---------------------------------------------------------------------------
// Reparameterization + per-row KL.  encoded[row] = [mu(256) | logvar(256)].
// ---------------------------------------------------------------------------
__global__ __launch_bounds__(256) void reparam_kl(
    const float* __restrict__ enc, const float* __restrict__ eps,
    __bf16* __restrict__ zb, float* __restrict__ kl)
{
  __shared__ float red[256];
  const int row = blockIdx.x, tid = threadIdx.x;
  const float mu = enc[(size_t)row * 512 + tid];
  const float lv = enc[(size_t)row * 512 + 256 + tid];
  const float z  = mu + expf(0.5f * lv) * eps[(size_t)row * CD + tid];
  zb[(size_t)row * CD + tid] = (__bf16)z;
  red[tid] = 1.0f + lv - mu * mu - expf(lv);
  __syncthreads();
  for (int s = 128; s; s >>= 1) { if (tid < s) red[tid] += red[tid + s]; __syncthreads(); }
  if (tid == 0) kl[row] = -0.5f * red[0];
}

// ---------------------------------------------------------------------------
// Per-row MSE vs target (len H); optionally also emit bf16 copy of pred.
// ---------------------------------------------------------------------------
__global__ __launch_bounds__(256) void mse_row(
    const float* __restrict__ pred, const float* __restrict__ tgt,
    __bf16* __restrict__ predb, float* __restrict__ loss)
{
  __shared__ float red[256];
  const int row = blockIdx.x, tid = threadIdx.x;
  float s = 0.0f;
  for (int j = tid; j < HD; j += 256) {
    const float p = pred[(size_t)row * HD + j];
    if (predb) predb[(size_t)row * HD + j] = (__bf16)p;
    const float d = p - tgt[(size_t)row * HD + j];
    s += d * d;
  }
  red[tid] = s; __syncthreads();
  for (int st = 128; st; st >>= 1) { if (tid < st) red[tid] += red[tid + st]; __syncthreads(); }
  if (tid == 0) loss[row] = red[0] * (1.0f / HD);
}

// ---------------------------------------------------------------------------
// Persistent GRU recurrence (batch=1), one grid barrier per timestep.
// gx[T,3H] precomputed (x@Wih^T + bih).  64 blocks; block owns j-chunk of 16
// and computes its r/z/n gate rows itself (48 dot products of length 1024,
// bf16 weights streamed from L2, 6 dots per wave, lane-strided + shfl reduce).
// h (4KB) is staged into LDS once per timestep right after the grid barrier's
// acquire fence, so the 48 dots hit LDS instead of global.
// Double-buffered h; hA must be zeroed before launch.
// ---------------------------------------------------------------------------
#define GRU_BLOCKS 64
#define JPB (HD / GRU_BLOCKS)   // 16

__global__ __launch_bounds__(256) void gru_recurrent(
    const float* __restrict__ gx, const __bf16* __restrict__ whh,
    const float* __restrict__ bhh, float* __restrict__ hA, float* __restrict__ hB,
    float* __restrict__ y, unsigned* __restrict__ cnt)
{
  __shared__ float gh_s[48];
  __shared__ float hs[HD];
  const int tid  = threadIdx.x;
  const int wave = tid >> 5, lane = tid & 31;
  const int j0   = blockIdx.x * JPB;

  for (int t = 0; t < NS; ++t) {
    const float* hc  = (t & 1) ? hB : hA;
    float*       hn  = (t & 1) ? hA : hB;
    const float* gxt = gx + (size_t)t * G3H;

    // stage h into LDS (256 threads x float4 = 1024 floats)
    reinterpret_cast<float4*>(hs)[tid] =
        reinterpret_cast<const float4*>(hc)[tid];
    __syncthreads();

    // 48 dots: gate g in {r,z,n}, jj in [0,16): row = g*H + j0 + jj
    for (int d = wave; d < 48; d += 8) {
      const int gate = d >> 4, jj = d & 15;
      const __bf16* wr = whh + (size_t)(gate * HD + j0 + jj) * HD;
      float s = 0.0f;
#pragma unroll 8
      for (int k = lane; k < HD; k += 32) s += (float)wr[k] * hs[k];
#pragma unroll
      for (int off = 16; off; off >>= 1) s += __shfl_xor(s, off, 32);
      if (lane == 0) gh_s[d] = s;
    }
    __syncthreads();

    if (tid < JPB) {
      const int j = j0 + tid;
      const float hr  = gh_s[tid]      + bhh[j];
      const float hz  = gh_s[16 + tid] + bhh[HD + j];
      const float hnn = gh_s[32 + tid] + bhh[2 * HD + j];
      const float xr = gxt[j], xz = gxt[HD + j], xn = gxt[2 * HD + j];
      const float rg = 1.0f / (1.0f + expf(-(xr + hr)));
      const float zg = 1.0f / (1.0f + expf(-(xz + hz)));
      const float ng = tanhf(xn + rg * hnn);
      const float hv = (1.0f - zg) * ng + zg * hs[j];
      hn[j] = hv;
      y[(size_t)t * HD + j] = hv;
    }
    __syncthreads();

    // grid-wide barrier (monotonic counter, zeroed by host each launch)
    if (tid == 0) {
      __threadfence();
      atomicAdd(cnt, 1u);
      const unsigned target = (unsigned)(t + 1) * gridDim.x;
      while (*(volatile unsigned*)cnt < target) __builtin_amdgcn_s_sleep(2);
    }
    __syncthreads();
    __threadfence();   // acquire: drop stale h lines from WGP caches
  }
}

// ---------------------------------------------------------------------------
// Final reduction + importance update
// ---------------------------------------------------------------------------
__global__ void copy_importance(const float* __restrict__ src, float* __restrict__ dst, int n)
{
  for (int i = blockIdx.x * blockDim.x + threadIdx.x; i < n;
       i += gridDim.x * blockDim.x)
    dst[i] = src[i];
}

__global__ __launch_bounds__(256) void total_err_kernel(
    const float* __restrict__ rl, const float* __restrict__ wl,
    const float* __restrict__ kl, float* __restrict__ te, float* __restrict__ mean_out)
{
  __shared__ float red[256];
  const int i = blockIdx.x * 256 + threadIdx.x;
  const float v = rl[i] + 0.1f * wl[i] + 0.001f * kl[i];
  te[i] = v;
  red[threadIdx.x] = v; __syncthreads();
  for (int s = 128; s; s >>= 1) {
    if (threadIdx.x < s) red[threadIdx.x] += red[threadIdx.x + s];
    __syncthreads();
  }
  if (threadIdx.x == 0) atomicAdd(mean_out, red[0] * (1.0f / NS));
}

__global__ void scatter_importance(const int* __restrict__ idx,
                                   const float* __restrict__ imp,
                                   const float* __restrict__ te,
                                   float* __restrict__ dst)
{
  const int i = blockIdx.x * blockDim.x + threadIdx.x;
  if (i < NS) {
    const int j = idx[i];
    dst[j] = 0.9f * imp[j] + 0.1f * te[i];
  }
}

// ---------------------------------------------------------------------------
// Host driver
// ---------------------------------------------------------------------------
extern "C" void kernel_launch(void* const* d_in, const int* in_sizes, int n_in,
                              void* d_out, int out_size, void* d_ws, size_t ws_size,
                              hipStream_t stream) {
  (void)in_sizes; (void)n_in; (void)out_size; (void)ws_size;

  const float* ep_mem   = (const float*)d_in[0];
  const float* mem_imp  = (const float*)d_in[1];
  const float* eps      = (const float*)d_in[2];
  const int*   idx      = (const int*)  d_in[3];
  const float* enc_w1   = (const float*)d_in[4];
  const float* enc_b1   = (const float*)d_in[5];
  const float* enc_g    = (const float*)d_in[6];
  const float* enc_beta = (const float*)d_in[7];
  const float* enc_w2   = (const float*)d_in[8];
  const float* enc_b2   = (const float*)d_in[9];
  const float* dec_w1   = (const float*)d_in[10];
  const float* dec_b1   = (const float*)d_in[11];
  const float* dec_g    = (const float*)d_in[12];
  const float* dec_beta = (const float*)d_in[13];
  const float* dec_w2   = (const float*)d_in[14];
  const float* dec_b2   = (const float*)d_in[15];
  const float* gwih0    = (const float*)d_in[16];
  const float* gwhh0    = (const float*)d_in[17];
  const float* gbih0    = (const float*)d_in[18];
  const float* gbhh0    = (const float*)d_in[19];
  const float* gwih1    = (const float*)d_in[20];
  const float* gwhh1    = (const float*)d_in[21];
  const float* gbih1    = (const float*)d_in[22];
  const float* gbhh1    = (const float*)d_in[23];
  float* out = (float*)d_out;     // out[0]=mean total_error, out[1..MMEM]=importance

  // ---- workspace layout --------------------------------------------------
  char* ws = (char*)d_ws;
  size_t off = 0;
  auto alloc = [&](size_t bytes) -> char* {
    char* p = ws + off;
    off = (off + bytes + 255) & ~(size_t)255;
    return p;
  };
  __bf16* wE1b  = (__bf16*)alloc((size_t)512 * HD * 2);
  __bf16* wE2b  = (__bf16*)alloc((size_t)512 * 512 * 2);
  __bf16* wD1b  = (__bf16*)alloc((size_t)512 * CD * 2);
  __bf16* wD2b  = (__bf16*)alloc((size_t)HD * 512 * 2);
  __bf16* wIH0b = (__bf16*)alloc((size_t)G3H * HD * 2);
  __bf16* wHH0b = (__bf16*)alloc((size_t)G3H * HD * 2);
  __bf16* wIH1b = (__bf16*)alloc((size_t)G3H * HD * 2);
  __bf16* wHH1b = (__bf16*)alloc((size_t)G3H * HD * 2);
  float*  samples  = (float*) alloc((size_t)NS * HD * 4);
  __bf16* samplesb = (__bf16*)alloc((size_t)NS * HD * 2);
  float*  e1     = (float*) alloc((size_t)NS * 512 * 4);
  __bf16* hb     = (__bf16*)alloc((size_t)NS * 512 * 2);
  float*  enc    = (float*) alloc((size_t)NS * 512 * 4);
  __bf16* zb     = (__bf16*)alloc((size_t)NS * CD * 2);
  float*  klv    = (float*) alloc((size_t)NS * 4);
  float*  dd1    = (float*) alloc((size_t)NS * 512 * 4);
  __bf16* hdb    = (__bf16*)alloc((size_t)NS * 512 * 2);
  float*  recon  = (float*) alloc((size_t)NS * HD * 4);
  __bf16* reconb = (__bf16*)alloc((size_t)NS * HD * 2);
  float*  rloss  = (float*) alloc((size_t)NS * 4);
  float*  gx     = (float*) alloc((size_t)NS * G3H * 4);  // reused for both layers
  float*  y1     = (float*) alloc((size_t)NS * HD * 4);
  __bf16* y1b    = (__bf16*)alloc((size_t)NS * HD * 2);
  float*  world  = (float*) alloc((size_t)NS * HD * 4);
  float*  wloss  = (float*) alloc((size_t)NS * 4);
  float*  te     = (float*) alloc((size_t)NS * 4);
  float*  hA     = (float*) alloc((size_t)HD * 4);
  float*  hB     = (float*) alloc((size_t)HD * 4);
  unsigned* cnts = (unsigned*)alloc(256);                 // cnts[0], cnts[1]

  // ---- per-call init (graph-replay deterministic) ------------------------
  hipMemsetAsync(cnts, 0, 256, stream);
  hipMemsetAsync(out, 0, sizeof(float), stream);

  // ---- weights -> bf16 ---------------------------------------------------
  cvt_f32_bf16<<<512, 256, 0, stream>>>(enc_w1, wE1b, 512 * HD);
  cvt_f32_bf16<<<512, 256, 0, stream>>>(enc_w2, wE2b, 512 * 512);
  cvt_f32_bf16<<<512, 256, 0, stream>>>(dec_w1, wD1b, 512 * CD);
  cvt_f32_bf16<<<512, 256, 0, stream>>>(dec_w2, wD2b, HD * 512);
  cvt_f32_bf16<<<1024, 256, 0, stream>>>(gwih0, wIH0b, G3H * HD);
  cvt_f32_bf16<<<1024, 256, 0, stream>>>(gwhh0, wHH0b, G3H * HD);
  cvt_f32_bf16<<<1024, 256, 0, stream>>>(gwih1, wIH1b, G3H * HD);
  cvt_f32_bf16<<<1024, 256, 0, stream>>>(gwhh1, wHH1b, G3H * HD);

  // ---- gather ------------------------------------------------------------
  gather_rows<<<NS, 256, 0, stream>>>(ep_mem, idx, samples, samplesb);

  // ---- VAE encode --------------------------------------------------------
  gemm_bf16_nt_bias<<<dim3(512 / 64, NS / 128), 256, 0, stream>>>(
      samplesb, wE1b, enc_b1, e1, NS, 512, HD);
  ln_gelu_512<<<NS, 256, 0, stream>>>(e1, enc_g, enc_beta, hb);
  gemm_bf16_nt_bias<<<dim3(512 / 64, NS / 128), 256, 0, stream>>>(
      hb, wE2b, enc_b2, enc, NS, 512, 512);
  reparam_kl<<<NS, 256, 0, stream>>>(enc, eps, zb, klv);

  // ---- decode ------------------------------------------------------------
  gemm_bf16_nt_bias<<<dim3(512 / 64, NS / 128), 256, 0, stream>>>(
      zb, wD1b, dec_b1, dd1, NS, 512, CD);
  ln_gelu_512<<<NS, 256, 0, stream>>>(dd1, dec_g, dec_beta, hdb);
  gemm_bf16_nt_bias<<<dim3(HD / 64, NS / 128), 256, 0, stream>>>(
      hdb, wD2b, dec_b2, recon, NS, HD, 512);
  mse_row<<<NS, 256, 0, stream>>>(recon, samples, reconb, rloss);

  // ---- GRU layer 0 -------------------------------------------------------
  gemm_bf16_nt_bias<<<dim3(G3H / 64, NS / 128), 256, 0, stream>>>(
      reconb, wIH0b, gbih0, gx, NS, G3H, HD);
  hipMemsetAsync(hA, 0, HD * sizeof(float), stream);
  gru_recurrent<<<GRU_BLOCKS, 256, 0, stream>>>(gx, wHH0b, gbhh0, hA, hB, y1, &cnts[0]);

  // ---- GRU layer 1 -------------------------------------------------------
  cvt_f32_bf16<<<1024, 256, 0, stream>>>(y1, y1b, NS * HD);
  gemm_bf16_nt_bias<<<dim3(G3H / 64, NS / 128), 256, 0, stream>>>(
      y1b, wIH1b, gbih1, gx, NS, G3H, HD);
  hipMemsetAsync(hA, 0, HD * sizeof(float), stream);
  gru_recurrent<<<GRU_BLOCKS, 256, 0, stream>>>(gx, wHH1b, gbhh1, hA, hB, world, &cnts[1]);

  // ---- losses + importance update ---------------------------------------
  mse_row<<<NS, 256, 0, stream>>>(world, samples, nullptr, wloss);
  copy_importance<<<512, 256, 0, stream>>>(mem_imp, out + 1, MMEM);
  total_err_kernel<<<NS / 256, 256, 0, stream>>>(rloss, wloss, klv, te, out);
  scatter_importance<<<NS / 256, 256, 0, stream>>>(idx, mem_imp, te, out + 1);
}
